// RasaswadayaGNN_26113401160011
// MI455X (gfx1250) — matrix operand order repaired and checked by reading, not compiled, not run on hardware
//
#include <hip/hip_runtime.h>
#include <hip/hip_bf16.h>
#include <stdint.h>

// ---------------- vector types ----------------
typedef __attribute__((ext_vector_type(16))) __bf16       v16bf;
typedef __attribute__((ext_vector_type(8)))  float        v8f;
typedef __attribute__((ext_vector_type(4)))  float        v4f;
typedef __attribute__((ext_vector_type(4)))  unsigned int u32x4;
typedef __attribute__((ext_vector_type(2)))  unsigned int u32x2;
typedef __attribute__((ext_vector_type(8)))  int          i32x8;
typedef __attribute__((ext_vector_type(4)))  int          i32x4;

union FragBF { v4f f4[2]; v16bf bf; };

__device__ __forceinline__ unsigned short f2bf(float f) {
  unsigned int u = __float_as_uint(f);
  u += 0x7FFFu + ((u >> 16) & 1u);      // round-to-nearest-even
  return (unsigned short)(u >> 16);
}

// ---------------- problem constants (from reference) ----------------
#define N_USERS 100000
#define N_ITEMS 100000
#define N_EDGES 600000
#define F_USER  96
#define F_ITEM  160
#define HID     128
#define OUTD    64

// ---------------- helpers ----------------
__global__ void fill_zero(float* __restrict__ p, int n) {
  int i = blockIdx.x * blockDim.x + threadIdx.x;
  int stride = gridDim.x * blockDim.x;
  for (; i < n; i += stride) p[i] = 0.0f;
}

// One-time weight prep: W [K,OUTC] f32 -> dst[n*KP + koff + k] bf16 (transposed,
// padded rows). This bakes the exact LDS image into global memory so per-block
// weight staging is a contiguous DMA.
__global__ void prep_wt(const float* __restrict__ W, int K, int OUTC,
                        unsigned short* __restrict__ dst, int KP, int koff) {
  int i = blockIdx.x * blockDim.x + threadIdx.x;
  int total = K * OUTC;
  int stride = gridDim.x * blockDim.x;
  for (; i < total; i += stride) {
    int k = i / OUTC, n = i - k * OUTC;
    dst[(size_t)n * KP + koff + k] = f2bf(W[i]);
  }
}

// one 128-thread block per edge: gather 128-dim row of src, atomic-add into dst
__global__ __launch_bounds__(128)
void scatter_accum(const float* __restrict__ feat,
                   const int* __restrict__ edges, int E,
                   float* __restrict__ agg, float* __restrict__ cnt) {
  int e = blockIdx.x;
  if (e >= E) return;
  int s = edges[e];          // row 0: src
  int d = edges[E + e];      // row 1: dst
  int f = threadIdx.x;
  atomicAdd(&agg[(size_t)d * HID + f], feat[(size_t)s * HID + f]);
  if (f == 0) atomicAdd(&cnt[d], 1.0f);
}

// ---------------- fused GEMM: Y = act( LN( A1*W1 + A2*W2 + b ) ) ----------------
// A1: [N,K1] f32 (optionally row-scaled by 1/max(cnt,1)); A2: [N,K2] f32 optional.
// WtG: pre-transposed bf16 weights, [OUTC][K1+K2+8] padded rows (the LDS image).
// gamma==nullptr -> no LayerNorm. Requires N % 32 == 0 (100000 = 32*3125).
template<int K1, int K2, int OUTC>
__global__ __launch_bounds__(OUTC * 2)
void gemm_bias_ln(const float* __restrict__ A1,
                  const float* __restrict__ rowcnt,
                  const float* __restrict__ A2,
                  const unsigned short* __restrict__ WtG,
                  const float* __restrict__ bias,
                  const float* __restrict__ gamma, const float* __restrict__ beta,
                  int relu, float* __restrict__ Y, int Nrows) {
  constexpr int KTOT   = K1 + K2;
  constexpr int ROWS   = 32;
  constexpr int NWAVES = OUTC / 16;
  constexpr int NTHR   = NWAVES * 32;
  constexpr int KP     = KTOT + 8;            // +16B pad: bank-conflict-free, 16B aligned
  static_assert(K1 % 32 == 0 || K2 == 0, "");
  static_assert(KTOT % 32 == 0 && K1 % 4 == 0 && (K2 % 4 == 0), "");
  static_assert((unsigned)OUTC * KP <= 65535u, "TDM tile_dim0 must fit in 16 bits");
  constexpr unsigned ABYTES = ROWS * KP * 2;
  constexpr unsigned YBYTES = ROWS * OUTC * 4;
  constexpr unsigned SBYTES = (ABYTES > YBYTES ? ABYTES : YBYTES);

  __shared__ __align__(16) unsigned short Wt[OUTC][KP];     // W^T bf16 (DMA target)
  __shared__ __align__(16) unsigned char  SBuf[SBYTES];     // As during GEMM, Ys for LN
  __shared__ float scaleS[ROWS];

  unsigned short (*As)[KP] = reinterpret_cast<unsigned short (*)[KP]>(SBuf);
  float (*Ys)[OUTC]        = reinterpret_cast<float (*)[OUTC]>(SBuf);

  const int tid = threadIdx.x;
  const int r0  = blockIdx.x * ROWS;
  if (r0 >= Nrows) return;

  // ---- kick off weight-tile DMA (TDM) from wave 0; contiguous 1-D tile ----
#if __has_builtin(__builtin_amdgcn_tensor_load_to_lds)
  if (tid < 32) {
    const unsigned NEL = (unsigned)(OUTC * KP);              // bf16 elements
    const unsigned ldsA = (unsigned)(size_t)(void*)&Wt[0][0]; // low 32b of flat = LDS offset
    const unsigned long long ga = (unsigned long long)WtG;
    u32x4 g0;
    g0[0] = 1u;                                              // count=1, no gather
    g0[1] = ldsA;                                            // lds_addr
    g0[2] = (unsigned)ga;                                    // global_addr[31:0]
    g0[3] = (unsigned)((ga >> 32) & 0x1FFFFFFu) | (2u << 30); // addr[56:32] | type=2
    i32x8 g1;
    g1[0] = (int)(1u << 16);                                 // wg_mask=0, data_size=2B
    g1[1] = (int)((NEL & 0xFFFFu) << 16);                    // tensor_dim0[15:0]
    g1[2] = (int)((NEL >> 16) | (1u << 16));                 // tensor_dim0[31:16], dim1=1
    g1[3] = (int)(NEL << 16);                                // tile_dim0 = NEL
    g1[4] = 1;                                               // tile_dim1 = 1
    g1[5] = (int)NEL;                                        // tensor_dim0_stride
    g1[6] = 0; g1[7] = 0;
    i32x4 z4 = {0, 0, 0, 0};
    i32x8 z8 = {0, 0, 0, 0, 0, 0, 0, 0};
    // 6-arg toolchain variant: (g0, g1, g2, g3, g4, cpol)
    __builtin_amdgcn_tensor_load_to_lds(g0, g1, z4, z4, z8, 0);
  }
#endif

  if (tid < ROWS) {
    float sc = 1.0f;
    if (rowcnt) sc = 1.0f / fmaxf(rowcnt[r0 + tid], 1.0f);
    scaleS[tid] = sc;
  }
  __syncthreads();

  // ---- stage A (f32 -> bf16, fold in mean-normalization); overlaps the DMA ----
  {
    constexpr int K1V = K1 / 4;
    const v4f* A1v = (const v4f*)A1;
    for (int idx = tid; idx < ROWS * K1V; idx += NTHR) {
      int row = idx / K1V, k4 = idx - row * K1V;
      v4f v = A1v[(size_t)(r0 + row) * K1V + k4];
      float sc = scaleS[row];
      u32x2 p;
      p[0] = (unsigned)f2bf(v[0] * sc) | ((unsigned)f2bf(v[1] * sc) << 16);
      p[1] = (unsigned)f2bf(v[2] * sc) | ((unsigned)f2bf(v[3] * sc) << 16);
      *(u32x2*)&As[row][k4 * 4] = p;
    }
  }
  if constexpr (K2 > 0) {
    constexpr int K2V = K2 / 4;
    const v4f* A2v = (const v4f*)A2;
    for (int idx = tid; idx < ROWS * K2V; idx += NTHR) {
      int row = idx / K2V, k4 = idx - row * K2V;
      v4f v = A2v[(size_t)(r0 + row) * K2V + k4];
      u32x2 p;
      p[0] = (unsigned)f2bf(v[0]) | ((unsigned)f2bf(v[1]) << 16);
      p[1] = (unsigned)f2bf(v[2]) | ((unsigned)f2bf(v[3]) << 16);
      *(u32x2*)&As[row][K1 + k4 * 4] = p;
    }
  }

#if __has_builtin(__builtin_amdgcn_tensor_load_to_lds)
  if (tid < 32) __builtin_amdgcn_s_wait_tensorcnt(0);        // wave 0 owns TENSORcnt
#else
  {
    // fallback: contiguous vectorized copy of the pre-built LDS image
    const u32x4* src = (const u32x4*)WtG;
    u32x4* dst = (u32x4*)&Wt[0][0];
    constexpr int NV = OUTC * KP / 8;
    for (int idx = tid; idx < NV; idx += NTHR) dst[idx] = src[idx];
  }
#endif
  __syncthreads();

  // ---- WMMA: each wave owns a 16-wide column strip, 2 row-tiles (32 rows) ----
  const int wave = tid >> 5;
  const int lane = tid & 31;
  const int hlf  = lane >> 4;
  const int lm   = lane & 15;
  const int n0   = wave * 16;

  v8f acc0 = {}; v8f acc1 = {};
  constexpr int KSTEPS = KTOT / 32;
  #pragma unroll
  for (int kk = 0; kk < KSTEPS; ++kk) {
    FragBF b;   // B fragment (32x16 bf16): K = {hlf*8..+7, 16+hlf*8..+7}
    const unsigned short* bp = &Wt[n0 + lm][kk * 32 + hlf * 8];
    b.f4[0] = *(const v4f*)(bp);
    b.f4[1] = *(const v4f*)(bp + 16);
    {
      FragBF a; // rows 0..15: K = {hlf*8..+7, 16+hlf*8..+7}
      const unsigned short* ap = &As[lm][kk * 32 + hlf * 8];
      a.f4[0] = *(const v4f*)(ap);
      a.f4[1] = *(const v4f*)(ap + 16);
      acc0 = __builtin_amdgcn_wmma_f32_16x16x32_bf16(false, a.bf, false, b.bf,
                                                     (short)0, acc0, false, false);
    }
    {
      FragBF a; // rows 16..31
      const unsigned short* ap = &As[16 + lm][kk * 32 + hlf * 8];
      a.f4[0] = *(const v4f*)(ap);
      a.f4[1] = *(const v4f*)(ap + 16);
      acc1 = __builtin_amdgcn_wmma_f32_16x16x32_bf16(false, a.bf, false, b.bf,
                                                     (short)0, acc1, false, false);
    }
  }

  const float bval = bias[n0 + lm];

  if (gamma == nullptr) {
    #pragma unroll
    for (int r = 0; r < 8; ++r) {
      int m = hlf * 8 + r;                      // C/D layout: M = r + half*8
      float v0 = acc0[r] + bval;
      float v1 = acc1[r] + bval;
      if (relu) { v0 = fmaxf(v0, 0.0f); v1 = fmaxf(v1, 0.0f); }
      Y[(size_t)(r0 + m) * OUTC + n0 + lm]      = v0;
      Y[(size_t)(r0 + 16 + m) * OUTC + n0 + lm] = v1;
    }
  } else {
    __syncthreads();                            // As lifetime ends; SBuf becomes Ys
    #pragma unroll
    for (int r = 0; r < 8; ++r) {
      int m = hlf * 8 + r;
      Ys[m][n0 + lm]      = acc0[r] + bval;
      Ys[16 + m][n0 + lm] = acc1[r] + bval;
    }
    __syncthreads();
    for (int row = wave; row < ROWS; row += NWAVES) {
      float s = 0.0f, s2 = 0.0f;
      for (int c = lane; c < OUTC; c += 32) {
        float v = Ys[row][c]; s += v; s2 += v * v;
      }
      #pragma unroll
      for (int off = 16; off > 0; off >>= 1) {
        s  += __shfl_xor(s,  off, 32);
        s2 += __shfl_xor(s2, off, 32);
      }
      float mean = s  * (1.0f / OUTC);
      float var  = s2 * (1.0f / OUTC) - mean * mean;
      float inv  = rsqrtf(var + 1e-5f);
      for (int c = lane; c < OUTC; c += 32) {
        float v = (Ys[row][c] - mean) * inv * gamma[c] + beta[c];
        if (relu) v = fmaxf(v, 0.0f);
        Y[(size_t)(r0 + row) * OUTC + c] = v;
      }
    }
  }
}

// ---------------- driver ----------------
extern "C" void kernel_launch(void* const* d_in, const int* in_sizes, int n_in,
                              void* d_out, int out_size, void* d_ws, size_t ws_size,
                              hipStream_t stream) {
  (void)in_sizes; (void)n_in; (void)out_size; (void)ws_size;
  const float* x_user  = (const float*)d_in[0];
  const float* x_item  = (const float*)d_in[1];
  const int*   edge_ui = (const int*)d_in[2];
  const int*   edge_iu = (const int*)d_in[3];
  const float* Wp_u = (const float*)d_in[4];   const float* bp_u = (const float*)d_in[5];
  const float* Wp_i = (const float*)d_in[6];   const float* bp_i = (const float*)d_in[7];
  const float* Wl0_ui = (const float*)d_in[8]; const float* bl0_ui = (const float*)d_in[9];
  const float* Wr0_ui = (const float*)d_in[10];
  const float* Wl0_iu = (const float*)d_in[11];const float* bl0_iu = (const float*)d_in[12];
  const float* Wr0_iu = (const float*)d_in[13];
  const float* g0_u = (const float*)d_in[14];  const float* b0_u = (const float*)d_in[15];
  const float* g0_i = (const float*)d_in[16];  const float* b0_i = (const float*)d_in[17];
  const float* Wl1_ui = (const float*)d_in[18];const float* bl1_ui = (const float*)d_in[19];
  const float* Wr1_ui = (const float*)d_in[20];
  const float* Wl1_iu = (const float*)d_in[21];const float* bl1_iu = (const float*)d_in[22];
  const float* Wr1_iu = (const float*)d_in[23];
  const float* g1_u = (const float*)d_in[24];  const float* b1_u = (const float*)d_in[25];
  const float* g1_i = (const float*)d_in[26];  const float* b1_i = (const float*)d_in[27];

  float* ws   = (float*)d_ws;
  float* h_u  = ws; ws += (size_t)N_USERS * HID;
  float* h_i  = ws; ws += (size_t)N_ITEMS * HID;
  float* h_u1 = ws; ws += (size_t)N_USERS * HID;
  float* h_i1 = ws; ws += (size_t)N_ITEMS * HID;
  float* agg  = ws; ws += (size_t)N_USERS * HID;   // reused for every edge type
  float* cnt  = ws; ws += N_USERS;

  // pre-transposed bf16 weight images (row stride = KTOT+8, 16B aligned)
  unsigned short* wtb   = (unsigned short*)ws;
  unsigned short* wtPu  = wtb; wtb += 128 * 104;   // [128][96+8]
  unsigned short* wtPi  = wtb; wtb += 128 * 168;   // [128][160+8]
  unsigned short* wt0ui = wtb; wtb += 128 * 264;   // [128][256+8]
  unsigned short* wt0iu = wtb; wtb += 128 * 264;
  unsigned short* wt1ui = wtb; wtb += 64 * 264;    // [64][256+8]
  unsigned short* wt1iu = wtb; wtb += 64 * 264;

  float* out_u = (float*)d_out;                               // h_u2 [N_USERS,64]
  float* out_i = (float*)d_out + (size_t)N_USERS * OUTD;      // h_i2 [N_ITEMS,64]

  const int gemmBlocks = N_USERS / 32;    // 3125, exact (100000 % 32 == 0)
  const int zB = 2048, zT = 256;

  // ---- one-time weight transpose/convert into the LDS image layout ----
  prep_wt<<<64, 256, 0, stream>>>(Wp_u,   F_USER, HID,  wtPu,  104, 0);
  prep_wt<<<64, 256, 0, stream>>>(Wp_i,   F_ITEM, HID,  wtPi,  168, 0);
  prep_wt<<<64, 256, 0, stream>>>(Wl0_ui, HID,    HID,  wt0ui, 264, 0);
  prep_wt<<<64, 256, 0, stream>>>(Wr0_ui, HID,    HID,  wt0ui, 264, HID);
  prep_wt<<<64, 256, 0, stream>>>(Wl0_iu, HID,    HID,  wt0iu, 264, 0);
  prep_wt<<<64, 256, 0, stream>>>(Wr0_iu, HID,    HID,  wt0iu, 264, HID);
  prep_wt<<<64, 256, 0, stream>>>(Wl1_ui, HID,    OUTD, wt1ui, 264, 0);
  prep_wt<<<64, 256, 0, stream>>>(Wr1_ui, HID,    OUTD, wt1ui, 264, HID);
  prep_wt<<<64, 256, 0, stream>>>(Wl1_iu, HID,    OUTD, wt1iu, 264, 0);
  prep_wt<<<64, 256, 0, stream>>>(Wr1_iu, HID,    OUTD, wt1iu, 264, HID);

  // ---- per-type input projections + ReLU ----
  gemm_bias_ln<F_USER, 0, HID><<<gemmBlocks, 256, 0, stream>>>(
      x_user, nullptr, nullptr, wtPu, bp_u, nullptr, nullptr, 1, h_u, N_USERS);
  gemm_bias_ln<F_ITEM, 0, HID><<<gemmBlocks, 256, 0, stream>>>(
      x_item, nullptr, nullptr, wtPi, bp_i, nullptr, nullptr, 1, h_i, N_ITEMS);

  // ---- layer 0, item side: mean(user -> item), combine, LN, ReLU ----
  fill_zero<<<zB, zT, 0, stream>>>(agg, N_ITEMS * HID);
  fill_zero<<<zB, zT, 0, stream>>>(cnt, N_ITEMS);
  scatter_accum<<<N_EDGES, 128, 0, stream>>>(h_u, edge_ui, N_EDGES, agg, cnt);
  gemm_bias_ln<HID, HID, HID><<<gemmBlocks, 256, 0, stream>>>(
      agg, cnt, h_i, wt0ui, bl0_ui, g0_i, b0_i, 1, h_i1, N_ITEMS);

  // ---- layer 0, user side ----
  fill_zero<<<zB, zT, 0, stream>>>(agg, N_USERS * HID);
  fill_zero<<<zB, zT, 0, stream>>>(cnt, N_USERS);
  scatter_accum<<<N_EDGES, 128, 0, stream>>>(h_i, edge_iu, N_EDGES, agg, cnt);
  gemm_bias_ln<HID, HID, HID><<<gemmBlocks, 256, 0, stream>>>(
      agg, cnt, h_u, wt0iu, bl0_iu, g0_u, b0_u, 1, h_u1, N_USERS);

  // ---- layer 1, item side: combine, LN (no ReLU) ----
  fill_zero<<<zB, zT, 0, stream>>>(agg, N_ITEMS * HID);
  fill_zero<<<zB, zT, 0, stream>>>(cnt, N_ITEMS);
  scatter_accum<<<N_EDGES, 128, 0, stream>>>(h_u1, edge_ui, N_EDGES, agg, cnt);
  gemm_bias_ln<HID, HID, OUTD><<<gemmBlocks, 128, 0, stream>>>(
      agg, cnt, h_i1, wt1ui, bl1_ui, g1_i, b1_i, 0, out_i, N_ITEMS);

  // ---- layer 1, user side ----
  fill_zero<<<zB, zT, 0, stream>>>(agg, N_USERS * HID);
  fill_zero<<<zB, zT, 0, stream>>>(cnt, N_USERS);
  scatter_accum<<<N_EDGES, 128, 0, stream>>>(h_i1, edge_iu, N_EDGES, agg, cnt);
  gemm_bias_ln<HID, HID, OUTD><<<gemmBlocks, 128, 0, stream>>>(
      agg, cnt, h_u1, wt1iu, bl1_iu, g1_u, b1_u, 0, out_u, N_USERS);
}
